// VectorQuantizer_57269093925315
// MI455X (gfx1250) — matrix operand order repaired and checked by reading, not compiled, not run on hardware
//
#include <hip/hip_runtime.h>
#include <hip/hip_bf16.h>
#include <math.h>

#define NROWS 32768
#define DDIM  256
#define KCB   8192
#define DECAYF 0.99f
#define OMDECAY 0.01f
#define COMMITF 0.25f
#define ENTRATIO 0.1f
#define EPSV 1e-5f

typedef __attribute__((ext_vector_type(16))) __bf16 v16bf;
typedef __attribute__((ext_vector_type(8)))  float v8f;
typedef __attribute__((ext_vector_type(4)))  unsigned int u32x4;
typedef __attribute__((ext_vector_type(8)))  unsigned int u32x8;

// ---- workspace layout (bytes) ----
#define O_XBF   0UL            // 32768*256*2
#define O_CBBF  16777216UL     // 8192*256*2
#define O_X2    20971520UL     // 32768*4
#define O_C2    21102592UL     // 8192*4
#define O_ROWM  21135360UL     // 32768*4
#define O_ROWLZ 21266432UL     // 32768*4
#define O_AVGP  21397504UL     // 8192*4
#define O_CNT   21430272UL     // 8192*4
#define O_DW    21463040UL     // 8192*256*4
#define O_SCAL  29851648UL     // 16 floats: 0 mse,1 rowterm,2 ae,3 pl,4 nsum

// ---- output layout (float elements) ----
#define OO_Q    0UL
#define OO_LOSS 8388608UL
#define OO_PERP 8388609UL
#define OO_IDX  8388610UL
#define OO_NCB  8421378UL
#define OO_NCS  10518530UL
#define OO_NCBE 10526722UL

// ---- CDNA5 async global->LDS copy (ASYNCcnt-tracked) ----
__device__ __forceinline__ void async_b128(unsigned lds_off, const char* gptr) {
  asm volatile("global_load_async_to_lds_b128 %0, %1, off"
               :: "v"(lds_off), "v"(gptr)
               : "memory");
}
__device__ __forceinline__ void wait_asynccnt0() {
  asm volatile("s_wait_asynccnt 0x0" ::: "memory");
}
__device__ __forceinline__ void wait_asynccnt4() {   // allow one 4-load tile in flight
  asm volatile("s_wait_asynccnt 0x4" ::: "memory");
}
__device__ __forceinline__ void wait_dscnt0() {
  asm volatile("s_wait_dscnt 0x0" ::: "memory");
}
__device__ __forceinline__ unsigned lds_off_of(const void* p) {
  return (unsigned)(uintptr_t)p;   // flat LDS addr[31:0] == wave-relative LDS offset
}

// ---- CDNA5 Tensor Data Mover: 1-D 64KB tile, D# groups 0/1 in SGPRs ----
__device__ __forceinline__ void tdm_load_to_lds(u32x4 g0, u32x8 g1) {
  asm volatile("tensor_load_to_lds %0, %1" :: "s"(g0), "s"(g1) : "memory");
}
// issue from ONE wave only (TDM ignores EXEC; per-wave issue)
__device__ __forceinline__ void tdm_stage_64k(unsigned lds_off, const void* gptr) {
  unsigned long long ga = (unsigned long long)(uintptr_t)gptr;
  u32x4 g0 = {1u,                                   // count=1 valid descriptor
              lds_off,                              // lds_addr
              (unsigned)(ga & 0xFFFFFFFFu),         // global_addr[31:0]
              (unsigned)((ga >> 32) & 0x01FFFFFFu) | 0x80000000u}; // [56:32] | type=2
  u32x8 g1;
  g1[0] = 3u << 16;      // workgroup_mask=0, data_size=3 (8-byte units)
  g1[1] = 8192u << 16;   // tensor_dim0[15:0]=8192 in [31:16]
  g1[2] = 1u << 16;      // tensor_dim0[31:16]=0, tensor_dim1=1
  g1[3] = 8192u << 16;   // tile_dim0=8192 (x 8B = 64KB)
  g1[4] = 1u;            // tile_dim1=1, tile_dim2=0
  g1[5] = 8192u;         // tensor_dim0_stride[31:0]
  g1[6] = 0u;
  g1[7] = 0u;
  tdm_load_to_lds(g0, g1);
  __builtin_amdgcn_s_wait_tensorcnt((short)0);
}

__global__ void k_init(float* z, long n) {
  long i = (long)blockIdx.x * blockDim.x + threadIdx.x;
  long stride = (long)gridDim.x * blockDim.x;
  for (; i < n; i += stride) z[i] = 0.0f;
}

__device__ __forceinline__ unsigned short f2bf(float f) {
  unsigned int u = __float_as_uint(f);
  u += 0x7FFFu + ((u >> 16) & 1u);           // round-to-nearest-even
  return (unsigned short)(u >> 16);
}

// one block (256 thr) per row: bf16 convert + squared-norm
__global__ void k_prep(const float* __restrict__ src,
                       unsigned short* __restrict__ bf,
                       float* __restrict__ sq) {
  int row = blockIdx.x, t = threadIdx.x;
  float v = src[(long)row * DDIM + t];
  bf[(long)row * DDIM + t] = f2bf(v);
  float pp = v * v;
  #pragma unroll
  for (int off = 16; off; off >>= 1) pp += __shfl_xor(pp, off, 32);
  __shared__ float red[8];
  if ((t & 31) == 0) red[t >> 5] = pp;
  __syncthreads();
  if (t == 0) {
    float s = 0.0f;
    #pragma unroll
    for (int i = 0; i < 8; ++i) s += red[i];
    sq[row] = s;
  }
}

// A fragments for this wave's 16 rows, all 256 dims, from LDS (contiguous k = 16*hf + s per lane half)
__device__ __forceinline__ void load_afrags(const u32x4* sx, int wave, int p, int hf,
                                            v16bf* afr) {
  int row_local = wave * 16 + p;
  #pragma unroll
  for (int kb8 = 0; kb8 < 8; ++kb8) {
    int i0 = row_local * 32 + kb8 * 4 + hf * 2;
    u32x4 c0 = sx[i0];
    u32x4 c1 = sx[i0 + 1];
    afr[kb8] = __builtin_bit_cast(v16bf,
        __builtin_shufflevector(c0, c1, 0, 1, 2, 3, 4, 5, 6, 7));
  }
}

__device__ __forceinline__ v16bf ld_bfrag(const u32x4* bb, int idx) {
  u32x4 lo = bb[idx];
  u32x4 hi = bb[idx + 1];
  return __builtin_bit_cast(v16bf,
      __builtin_shufflevector(lo, hi, 0, 1, 2, 3, 4, 5, 6, 7));
}

// 16 rows x 32 codes: 16x v_wmma_f32_16x16x32_bf16, B fragments software-pipelined
// one k-step ahead so DS waits overlap with the current WMMAs
__device__ __forceinline__ void gemm32_lds(const v16bf* afr, const u32x4* bb,
                                           int p, int hf, v8f& acc0, v8f& acc1) {
  int base0 = p * 32 + hf * 2;          // code_local = p
  int base1 = (16 + p) * 32 + hf * 2;   // code_local = 16+p
  v16bf b0 = ld_bfrag(bb, base0);
  v16bf b1 = ld_bfrag(bb, base1);
  #pragma unroll
  for (int kb8 = 0; kb8 < 8; ++kb8) {
    if (kb8 < 7) {
      v16bf n0 = ld_bfrag(bb, base0 + (kb8 + 1) * 4);
      v16bf n1 = ld_bfrag(bb, base1 + (kb8 + 1) * 4);
      acc0 = __builtin_amdgcn_wmma_f32_16x16x32_bf16(false, afr[kb8], false, b0,
                                                     (short)0, acc0, false, false);
      acc1 = __builtin_amdgcn_wmma_f32_16x16x32_bf16(false, afr[kb8], false, b1,
                                                     (short)0, acc1, false, false);
      b0 = n0; b1 = n1;
    } else {
      acc0 = __builtin_amdgcn_wmma_f32_16x16x32_bf16(false, afr[kb8], false, b0,
                                                     (short)0, acc0, false, false);
      acc1 = __builtin_amdgcn_wmma_f32_16x16x32_bf16(false, afr[kb8], false, b1,
                                                     (short)0, acc1, false, false);
    }
  }
}

// async-copy one 16KB B-tile (32 codes x 512B) into LDS, 4x b128 per thread
__device__ __forceinline__ void copy_b_tile(unsigned lds_base, const char* gbase, int tid) {
  #pragma unroll
  for (int j = 0; j < 4; ++j)
    async_b128(lds_base + tid * 16 + j * 4096, gbase + tid * 16 + j * 4096);
}

// Pass 1: distances via WMMA; per-row argmin + online softmax-entropy stats
__global__ __launch_bounds__(256, 1)
void k_pass1(const unsigned short* __restrict__ xbf,
             const unsigned short* __restrict__ cbbf,
             const float* __restrict__ x2, const float* __restrict__ c2,
             float* __restrict__ out, float* __restrict__ rowM,
             float* __restrict__ rowLZ, float* __restrict__ scal) {
  // 64KB: phase 1 = 128-row x-tile (TDM); phase 2 = B triple buffer (3 x 16KB)
  __shared__ u32x4 smem[4096];
  int tid = threadIdx.x, blk = blockIdx.x;
  unsigned smoff = lds_off_of(&smem[0]);
  const char* xg = (const char*)xbf + (long)blk * 65536;

  if (tid < 32) tdm_stage_64k(smoff, xg);   // wave 0 drives the TDM
  __syncthreads();

  int wave = tid >> 5, lane = tid & 31, p = lane & 15, hf = lane >> 4;
  int r0 = blk * 128 + wave * 16;
  v16bf afr[8];
  load_afrags(smem, wave, p, hf, afr);
  wait_dscnt0();             // afr ds_loads landed before smem is overwritten
  __syncthreads();

  float x2r[8];
  #pragma unroll
  for (int v = 0; v < 8; ++v) x2r[v] = x2[r0 + hf * 8 + v];

  float m[8], s[8], t[8], bd[8];
  int bi[8];
  #pragma unroll
  for (int v = 0; v < 8; ++v) { m[v] = -1e30f; s[v] = 0.f; t[v] = 0.f; bd[v] = 3.0e38f; bi[v] = 0; }

  const char* cbg = (const char*)cbbf;
  copy_b_tile(smoff, cbg, tid);                   // tile 0 -> buf 0
  copy_b_tile(smoff + 16384, cbg + 16384, tid);   // tile 1 -> buf 1
  wait_asynccnt4();                               // tile 0 complete
  __syncthreads();

  const int NIT = KCB / 32;
  for (int it = 0; it < NIT; ++it) {
    int c0 = it * 32;
    const u32x4* bb = smem + (it % 3) * 1024;
    if (it + 2 < NIT)                             // prefetch depth 2
      copy_b_tile(smoff + ((it + 2) % 3) * 16384, cbg + (long)(it + 2) * 16384, tid);

    v8f acc0 = {0.f,0.f,0.f,0.f,0.f,0.f,0.f,0.f};
    v8f acc1 = {0.f,0.f,0.f,0.f,0.f,0.f,0.f,0.f};
    gemm32_lds(afr, bb, p, hf, acc0, acc1);
    float c20 = c2[c0 + p], c21 = c2[c0 + 16 + p];
    #pragma unroll
    for (int v = 0; v < 8; ++v) {
      float dist0 = sqrtf(fmaxf(x2r[v] - 2.0f * acc0[v] + c20, 0.0f));
      float dist1 = sqrtf(fmaxf(x2r[v] - 2.0f * acc1[v] + c21, 0.0f));
      if (dist0 < bd[v]) { bd[v] = dist0; bi[v] = c0 + p; }
      if (dist1 < bd[v]) { bd[v] = dist1; bi[v] = c0 + 16 + p; }
      float a0 = -dist0;
      if (a0 > m[v]) {                    // rare rescale path
        float sc = __expf(m[v] - a0);
        s[v] = s[v] * sc + 1.0f; t[v] = t[v] * sc + a0; m[v] = a0;
      } else {
        float e = __expf(a0 - m[v]);
        s[v] += e; t[v] += e * a0;
      }
      float a1 = -dist1;
      if (a1 > m[v]) {
        float sc = __expf(m[v] - a1);
        s[v] = s[v] * sc + 1.0f; t[v] = t[v] * sc + a1; m[v] = a1;
      } else {
        float e = __expf(a1 - m[v]);
        s[v] += e; t[v] += e * a1;
      }
    }
    if (it + 2 < NIT) wait_asynccnt4(); else wait_asynccnt0();
    __syncthreads();                              // everyone done reading bb
  }

  // merge across the 16 lanes of each wave half (C/D layout: lane = N, half = M+8)
  #pragma unroll
  for (int off = 8; off; off >>= 1) {
    #pragma unroll
    for (int v = 0; v < 8; ++v) {
      float om = __shfl_xor(m[v], off, 32);
      float os = __shfl_xor(s[v], off, 32);
      float ot = __shfl_xor(t[v], off, 32);
      float mn = fmaxf(m[v], om);
      float e0 = __expf(m[v] - mn), e1 = __expf(om - mn);
      s[v] = s[v] * e0 + os * e1;
      t[v] = t[v] * e0 + ot * e1;
      m[v] = mn;
      float obd = __shfl_xor(bd[v], off, 32);
      int   obi = __shfl_xor(bi[v], off, 32);
      if (obd < bd[v] || (obd == bd[v] && obi < bi[v])) { bd[v] = obd; bi[v] = obi; }
    }
  }
  if (p == 0) {
    float rts = 0.0f;
    #pragma unroll
    for (int v = 0; v < 8; ++v) {
      int row = r0 + hf * 8 + v;
      out[OO_IDX + row] = (float)bi[v];
      float lz = m[v] + __logf(s[v]);
      rowM[row] = m[v];
      rowLZ[row] = lz;
      rts += t[v] / s[v] - lz;   // sum_k p*log p for this row
    }
    atomicAdd(&scal[1], rts);
  }
}

// Pass 2: recompute distances via WMMA, accumulate avg_probs[k]
__global__ __launch_bounds__(256, 1)
void k_pass2(const unsigned short* __restrict__ xbf,
             const unsigned short* __restrict__ cbbf,
             const float* __restrict__ x2, const float* __restrict__ c2,
             const float* __restrict__ rowLZ, float* __restrict__ avgp) {
  __shared__ u32x4 smem[4096];
  int tid = threadIdx.x, blk = blockIdx.x;
  unsigned smoff = lds_off_of(&smem[0]);
  const char* xg = (const char*)xbf + (long)blk * 65536;

  if (tid < 32) tdm_stage_64k(smoff, xg);
  __syncthreads();

  int wave = tid >> 5, lane = tid & 31, p = lane & 15, hf = lane >> 4;
  int r0 = blk * 128 + wave * 16;
  v16bf afr[8];
  load_afrags(smem, wave, p, hf, afr);
  wait_dscnt0();
  __syncthreads();

  float x2r[8], lz[8];
  #pragma unroll
  for (int v = 0; v < 8; ++v) {
    x2r[v] = x2[r0 + hf * 8 + v];
    lz[v]  = rowLZ[r0 + hf * 8 + v];
  }

  const char* cbg = (const char*)cbbf;
  copy_b_tile(smoff, cbg, tid);
  copy_b_tile(smoff + 16384, cbg + 16384, tid);
  wait_asynccnt4();
  __syncthreads();

  const float invN = 1.0f / (float)NROWS;
  const int NIT = KCB / 32;
  for (int it = 0; it < NIT; ++it) {
    int c0 = it * 32;
    const u32x4* bb = smem + (it % 3) * 1024;
    if (it + 2 < NIT)
      copy_b_tile(smoff + ((it + 2) % 3) * 16384, cbg + (long)(it + 2) * 16384, tid);

    v8f acc0 = {0.f,0.f,0.f,0.f,0.f,0.f,0.f,0.f};
    v8f acc1 = {0.f,0.f,0.f,0.f,0.f,0.f,0.f,0.f};
    gemm32_lds(afr, bb, p, hf, acc0, acc1);
    float c20 = c2[c0 + p], c21 = c2[c0 + 16 + p];
    float ps0 = 0.0f, ps1 = 0.0f;
    #pragma unroll
    for (int v = 0; v < 8; ++v) {
      float dist0 = sqrtf(fmaxf(x2r[v] - 2.0f * acc0[v] + c20, 0.0f));
      float dist1 = sqrtf(fmaxf(x2r[v] - 2.0f * acc1[v] + c21, 0.0f));
      ps0 += __expf(-dist0 - lz[v]);
      ps1 += __expf(-dist1 - lz[v]);
    }
    ps0 += __shfl_xor(ps0, 16, 32);  // lanes L and L+16 hold the same column
    ps1 += __shfl_xor(ps1, 16, 32);
    if (hf == 0) {
      atomicAdd(&avgp[c0 + p], ps0 * invN);
      atomicAdd(&avgp[c0 + 16 + p], ps1 * invN);
    }
    if (it + 2 < NIT) wait_asynccnt4(); else wait_asynccnt0();
    __syncthreads();
  }
}

// gather quantized rows, mse, counts, dw segment-sum
__global__ void k_quant(const float* __restrict__ x, const float* __restrict__ cb,
                        const float* __restrict__ idxf, float* __restrict__ outq,
                        float* __restrict__ cnt, float* __restrict__ dw,
                        float* __restrict__ scal) {
  int row = blockIdx.x, t = threadIdx.x;
  int idx = (int)idxf[row];
  float q = cb[(long)idx * DDIM + t];
  float xv = x[(long)row * DDIM + t];
  outq[(long)row * DDIM + t] = q;   // straight-through value == quantized
  atomicAdd(&dw[(long)idx * DDIM + t], xv);
  float diff = q - xv;
  float p2 = diff * diff;
  #pragma unroll
  for (int off = 16; off; off >>= 1) p2 += __shfl_xor(p2, off, 32);
  if ((t & 31) == 0) atomicAdd(&scal[0], p2);
  if (t == 0) atomicAdd(&cnt[idx], 1.0f);
}

// per-codebook-entry stats + EMA cluster sizes
__global__ void k_stats(const float* __restrict__ avgp, const float* __restrict__ cnt,
                        const float* __restrict__ cse, float* __restrict__ out_ncs,
                        float* __restrict__ scal) {
  int k = blockIdx.x * 256 + threadIdx.x;
  float ap = avgp[k];
  float ae = ap * __logf(ap + 1e-5f);
  float apc = cnt[k] * (1.0f / (float)NROWS);
  float pl = apc * __logf(apc + 1e-10f);
  float ncs = DECAYF * cse[k] + OMDECAY * cnt[k];
  out_ncs[k] = ncs;
  float ns = ncs;
  #pragma unroll
  for (int off = 16; off; off >>= 1) {
    ae += __shfl_xor(ae, off, 32);
    pl += __shfl_xor(pl, off, 32);
    ns += __shfl_xor(ns, off, 32);
  }
  if ((threadIdx.x & 31) == 0) {
    atomicAdd(&scal[2], ae);
    atomicAdd(&scal[3], pl);
    atomicAdd(&scal[4], ns);
  }
}

__global__ void k_final(const float* __restrict__ scal, float* __restrict__ out) {
  if (threadIdx.x == 0 && blockIdx.x == 0) {
    float mse_sum = scal[0];
    float sample_ent = -scal[1] * (1.0f / (float)NROWS);
    float avg_ent = -scal[2];
    float loss = (1.0f + COMMITF) * mse_sum / ((float)NROWS * (float)DDIM)
               + ENTRATIO * (sample_ent - avg_ent);
    out[OO_LOSS] = loss;
    out[OO_PERP] = __expf(-scal[3]);
  }
}

__global__ void k_ema(const float* __restrict__ cbema, const float* __restrict__ dw,
                      const float* __restrict__ ncs_arr, const float* __restrict__ scal,
                      float* __restrict__ out_ncb, float* __restrict__ out_ncbe) {
  long i = (long)blockIdx.x * blockDim.x + threadIdx.x;
  int k = (int)(i >> 8);
  float ncbe = DECAYF * cbema[i] + OMDECAY * dw[i];
  out_ncbe[i] = ncbe;
  float n = scal[4];
  float ncs = ncs_arr[k];
  float smoothed = (ncs + EPSV) / (n + (float)KCB * EPSV) * n;
  out_ncb[i] = ncbe / smoothed;
}

extern "C" void kernel_launch(void* const* d_in, const int* in_sizes, int n_in,
                              void* d_out, int out_size, void* d_ws, size_t ws_size,
                              hipStream_t stream) {
  (void)in_sizes; (void)n_in; (void)out_size; (void)ws_size;
  const float* x     = (const float*)d_in[0];
  const float* cb    = (const float*)d_in[1];
  const float* cse   = (const float*)d_in[2];
  const float* cbema = (const float*)d_in[3];
  float* out = (float*)d_out;
  char* ws = (char*)d_ws;

  unsigned short* xbf  = (unsigned short*)(ws + O_XBF);
  unsigned short* cbbf = (unsigned short*)(ws + O_CBBF);
  float* x2    = (float*)(ws + O_X2);
  float* c2    = (float*)(ws + O_C2);
  float* rowM  = (float*)(ws + O_ROWM);
  float* rowLZ = (float*)(ws + O_ROWLZ);
  float* avgp  = (float*)(ws + O_AVGP);
  float* cnt   = (float*)(ws + O_CNT);
  float* dw    = (float*)(ws + O_DW);
  float* scal  = (float*)(ws + O_SCAL);

  long nz = (long)((O_SCAL + 64UL - O_AVGP) / 4UL);
  k_init<<<2048, 256, 0, stream>>>((float*)(ws + O_AVGP), nz);
  k_prep<<<NROWS, 256, 0, stream>>>(x, xbf, x2);
  k_prep<<<KCB, 256, 0, stream>>>(cb, cbbf, c2);
  k_pass1<<<NROWS / 128, 256, 0, stream>>>(xbf, cbbf, x2, c2, out, rowM, rowLZ, scal);
  k_quant<<<NROWS, 256, 0, stream>>>(x, cb, out + OO_IDX, out + OO_Q, cnt, dw, scal);
  k_pass2<<<NROWS / 128, 256, 0, stream>>>(xbf, cbbf, x2, c2, rowLZ, avgp);
  k_stats<<<KCB / 256, 256, 0, stream>>>(avgp, cnt, cse, out + OO_NCS, scal);
  k_final<<<1, 64, 0, stream>>>(scal, out);
  k_ema<<<(KCB * DDIM) / 256, 256, 0, stream>>>(cbema, dw, out + OO_NCS, scal,
                                                out + OO_NCB, out + OO_NCBE);
}
